// Attention_59227599012177
// MI455X (gfx1250) — matrix operand order repaired and checked by reading, not compiled, not run on hardware
//
#include <hip/hip_runtime.h>
#include <hip/hip_bf16.h>

// ---------------------------------------------------------------------------
// Attention block for MI455X (gfx1250, wave32, WMMA + async LDS pipelines).
// B=2, L=2048, DIM=2048, H=32, KV=8, HD=64, N_REP=4
// ---------------------------------------------------------------------------

#define B_  2
#define L_  2048
#define DIM_ 2048
#define H_  32
#define KV_ 8
#define HD_ 64

typedef __attribute__((ext_vector_type(16))) _Float16 v16h;
typedef __attribute__((ext_vector_type(8)))  _Float16 v8h;
typedef __attribute__((ext_vector_type(8)))  float    v8f;

// ---------------------------------------------------------------------------
// Fragment loaders matching CDNA5 WMMA 16-bit VGPR layouts (ISA 7.12.2).
// ---------------------------------------------------------------------------
__device__ __forceinline__ v16h load_a_frag(const _Float16* base,
                                            int ld, int row0, int k0, int lane) {
  int m  = lane & 15;
  int kb = k0 + ((lane >> 4) << 3);
  const _Float16* p = base + (size_t)(row0 + m) * ld + kb;
  v8h lo = *(const v8h*)(p);
  v8h hi = *(const v8h*)(p + 16);
  v16h a;
#pragma unroll
  for (int i = 0; i < 8; ++i) { a[i] = lo[i]; a[i + 8] = hi[i]; }
  return a;
}

__device__ __forceinline__ v16h load_b_frag(const _Float16* base,
                                            int ld, int col0, int k0, int lane) {
  int n  = lane & 15;
  int kb = k0 + ((lane >> 4) << 3);
  const _Float16* p = base + (size_t)(col0 + n) * ld + kb;
  v8h lo = *(const v8h*)(p);
  v8h hi = *(const v8h*)(p + 16);
  v16h b;
#pragma unroll
  for (int i = 0; i < 8; ++i) { b[i] = lo[i]; b[i + 8] = hi[i]; }
  return b;
}

__device__ __forceinline__ v8f wmma_f16(v16h a, v16h b, v8f c) {
  return __builtin_amdgcn_wmma_f32_16x16x32_f16(false, a, false, b,
                                                (short)0, c, false, false);
}

// ---------------------------------------------------------------------------
// CDNA5 async global -> LDS copy (ASYNCcnt, ISA 10.7 / 15.18.3).
// ---------------------------------------------------------------------------
__device__ __forceinline__ void async_b128(uint32_t lds_addr, const void* src) {
  asm volatile("global_load_async_to_lds_b128 %0, %1, off"
               :: "v"(lds_addr), "v"((uint64_t)(uintptr_t)src) : "memory");
}
__device__ __forceinline__ void wait_async_keep16() {
  asm volatile("s_wait_asynccnt 0x10" ::: "memory");
}
__device__ __forceinline__ void wait_async_keep4() {
  asm volatile("s_wait_asynccnt 0x4" ::: "memory");
}
__device__ __forceinline__ void wait_async_zero() {
  asm volatile("s_wait_asynccnt 0x0" ::: "memory");
}

// ---------------------------------------------------------------------------
// fp32 -> fp16 conversion
// ---------------------------------------------------------------------------
__global__ void cvt_f32_f16(const float* __restrict__ in,
                            _Float16* __restrict__ out, int n) {
  int i = blockIdx.x * 256 + threadIdx.x;
  if (i < n) out[i] = (_Float16)in[i];
}

// ---------------------------------------------------------------------------
// WMMA GEMM:  C[M,N] = A[M,K] * W[N,K]^T
// Block: 128 threads = 4 waves (2x2), block tile 128x128, wave tile 64x64.
// 128x64 k-slices of A and W double-buffered in LDS by async DMA:
// 32 WMMAs + 32 ds_load_b128 per barrier pair.
// mode 0: Q [b][h][l][d] | 1: K [b][kv][l][d] | 2: Vt [b][kv][d][l] | 3: fp32
// ---------------------------------------------------------------------------
__global__ __launch_bounds__(128)
void gemm_wmma(const _Float16* __restrict__ A, const _Float16* __restrict__ W,
               _Float16* __restrict__ outH, float* __restrict__ outF,
               int M, int N, int K, int mode) {
  __shared__ _Float16 Abuf[2][128 * 64];
  __shared__ _Float16 Bbuf[2][128 * 64];

  int tid  = threadIdx.x;
  int lane = tid & 31;
  int wave = tid >> 5;
  int wm = wave >> 1, wn = wave & 1;
  int rowBase = blockIdx.y * 128;
  int colBase = blockIdx.x * 128;

  v8f acc[4][4];
#pragma unroll
  for (int i = 0; i < 4; ++i)
#pragma unroll
    for (int j = 0; j < 4; ++j) acc[i][j] = (v8f){};

  const int steps = K >> 6;   // K / 64

  // Stage a 128x64 slice of A and of W (16 KB each, 16 B chunks).
  auto stage = [&](int k0, int buf) {
#pragma unroll
    for (int i = 0; i < 8; ++i) {
      int c   = tid + i * 128;
      int row = c >> 3;
      int off = (c & 7) * 8;            // halves within the 64-wide slice
      const _Float16* srcA = A + (size_t)(rowBase + row) * K + k0 + off;
      const _Float16* srcB = W + (size_t)(colBase + row) * K + k0 + off;
      async_b128((uint32_t)(uintptr_t)(&Abuf[buf][c * 8]), srcA);
      async_b128((uint32_t)(uintptr_t)(&Bbuf[buf][c * 8]), srcB);
    }
  };

  stage(0, 0);
  for (int it = 0; it < steps; ++it) {
    int buf = it & 1;
    bool more = (it + 1) < steps;
    if (more) stage((it + 1) << 6, buf ^ 1);
    if (more) wait_async_keep16(); else wait_async_zero();
    __syncthreads();

#pragma unroll
    for (int kk = 0; kk < 64; kk += 32) {
      v16h bf[4];
#pragma unroll
      for (int j = 0; j < 4; ++j)
        bf[j] = load_b_frag(&Bbuf[buf][0], 64, wn * 64 + j * 16, kk, lane);
#pragma unroll
      for (int i = 0; i < 4; ++i) {
        v16h af = load_a_frag(&Abuf[buf][0], 64, wm * 64 + i * 16, kk, lane);
#pragma unroll
        for (int j = 0; j < 4; ++j)
          acc[i][j] = wmma_f16(af, bf[j], acc[i][j]);
      }
    }
    __syncthreads();   // all reads done before buf is overwritten
  }

  // C layout: VGPR r, lanes 0-15 -> M=r, lanes 16-31 -> M=8+r; N = lane&15.
  int mloc = (lane >> 4) << 3;
  int n    = lane & 15;
#pragma unroll
  for (int i = 0; i < 4; ++i) {
#pragma unroll
    for (int j = 0; j < 4; ++j) {
#pragma unroll
      for (int r = 0; r < 8; ++r) {
        int row = rowBase + wm * 64 + i * 16 + mloc + r;
        int col = colBase + wn * 64 + j * 16 + n;
        float v = acc[i][j][r];
        if (mode == 3) {
          outF[(size_t)row * N + col] = v;
        } else {
          int b  = row >> 11;
          int l  = row & 2047;
          int hh = col >> 6;
          int d  = col & 63;
          size_t idx;
          if (mode == 0)      idx = (((size_t)(b * H_  + hh) * L_) + l) * HD_ + d;
          else if (mode == 1) idx = (((size_t)(b * KV_ + hh) * L_) + l) * HD_ + d;
          else                idx = (((size_t)(b * KV_ + hh) * HD_) + d) * L_ + l;
          outH[idx] = (_Float16)v;
        }
      }
    }
  }
}

// ---------------------------------------------------------------------------
// RoPE applied in place to f16 tensor [b*nh][L][HD]; freqs fp32 [2][L][HD/2].
// ---------------------------------------------------------------------------
__global__ void rope_kernel(_Float16* __restrict__ t,
                            const float* __restrict__ fc, int total) {
  int tid = blockIdx.x * 256 + threadIdx.x;
  if (tid >= total) return;
  int i  = tid & 31;
  int l  = (tid >> 5) & 2047;
  int bh = tid >> 16;
  float c = fc[l * 32 + i];
  float s = fc[L_ * 32 + l * 32 + i];
  _Float16* p = t + (((size_t)bh * L_) + l) * HD_ + 2 * i;
  float xr = (float)p[0];
  float xi = (float)p[1];
  p[0] = (_Float16)(xr * c - xi * s);
  p[1] = (_Float16)(xr * s + xi * c);
}

// ---------------------------------------------------------------------------
// Flash attention, causal, GQA — cooperative block version.
// grid: (L/128, B*H), block 256 = 8 waves. All waves share one (b,h); the
// block double-buffers the common 64x64 K and V tiles in LDS via async DMA.
// Wave w owns q rows [bx*128 + w*16, +16); computes while j0 <= qrow0+15
// (wave-uniform => EXEC all-ones for WMMA), always participates in barriers.
// ---------------------------------------------------------------------------
__global__ __launch_bounds__(256)
void flash_attn(const _Float16* __restrict__ Q, const _Float16* __restrict__ Kc,
                const _Float16* __restrict__ Vt, _Float16* __restrict__ O) {
  __shared__ _Float16 Kbuf[2][64 * 64];   // [key][d]
  __shared__ _Float16 Vbuf[2][64 * 64];   // [d][key]
  __shared__ _Float16 Pbuf[8][16 * 64];   // per-wave P transpose staging

  int tid   = threadIdx.x;
  int lane  = tid & 31;
  int wave  = tid >> 5;
  int qrow0 = blockIdx.x * 128 + wave * 16;
  int bh = blockIdx.y;
  int b  = bh >> 5;
  int h  = bh & 31;
  int kvh = h >> 2;

  const _Float16* Qb = Q  + ((size_t)bh * L_) * HD_;
  const _Float16* Kb = Kc + ((size_t)(b * KV_ + kvh) * L_) * HD_;
  const _Float16* Vb = Vt + ((size_t)(b * KV_ + kvh) * HD_) * L_;

  v16h qf0 = load_a_frag(Qb, HD_, qrow0, 0, lane);
  v16h qf1 = load_a_frag(Qb, HD_, qrow0, 32, lane);

  v8f o[4];
#pragma unroll
  for (int t = 0; t < 4; ++t) o[t] = (v8f){};
  float m_run[8], l_run[8];
#pragma unroll
  for (int r = 0; r < 8; ++r) { m_run[r] = -3.0e38f; l_run[r] = 0.0f; }

  int mloc = (lane >> 4) << 3;
  int ncol = lane & 15;
  const float SC = 0.125f * 1.44269504088896340736f;  // scale * log2(e)

  // Stage shared K tile (rows j0..j0+63 of Kb) and V tile (Vt cols j0..j0+63).
  auto stage = [&](int j0, int buf) {
#pragma unroll
    for (int i = 0; i < 2; ++i) {
      int c   = tid + i * 256;          // 512 chunks of 16 B per tile
      int row = c >> 3;
      int off = (c & 7) * 8;
      const _Float16* srcK = Kb + (size_t)(j0 + row) * HD_ + off;
      const _Float16* srcV = Vb + (size_t)row * L_ + j0 + off;
      async_b128((uint32_t)(uintptr_t)(&Kbuf[buf][c * 8]), srcK);
      async_b128((uint32_t)(uintptr_t)(&Vbuf[buf][c * 8]), srcV);
    }
  };

  const int nsteps = 2 * blockIdx.x + 2;   // keys 0 .. bx*128+127
  stage(0, 0);
  for (int it = 0; it < nsteps; ++it) {
    int j0  = it * 64;
    int buf = it & 1;
    bool more = (it + 1) < nsteps;
    if (more) stage((it + 1) * 64, buf ^ 1);
    if (more) wait_async_keep4(); else wait_async_zero();
    __syncthreads();

    if (j0 <= qrow0 + 15) {   // wave-uniform: EXEC stays all-ones inside
      // ---- S = Q K^T over the shared 64-key tile ----
      v8f s[4];
#pragma unroll
      for (int t = 0; t < 4; ++t) {
        s[t] = (v8f){};
        v16h kb0 = load_b_frag(&Kbuf[buf][0], HD_, t * 16, 0, lane);
        v16h kb1 = load_b_frag(&Kbuf[buf][0], HD_, t * 16, 32, lane);
        s[t] = wmma_f16(qf0, kb0, s[t]);
        s[t] = wmma_f16(qf1, kb1, s[t]);
      }
      // scale into log2 domain + causal mask
#pragma unroll
      for (int t = 0; t < 4; ++t) {
#pragma unroll
        for (int r = 0; r < 8; ++r) {
          float v = s[t][r] * SC;
          int key = j0 + t * 16 + ncol;
          int qq  = qrow0 + mloc + r;
          s[t][r] = (key > qq) ? -3.0e38f : v;
        }
      }
      // ---- online softmax (base 2) ----
#pragma unroll
      for (int r = 0; r < 8; ++r) {
        float rm = fmaxf(fmaxf(s[0][r], s[1][r]), fmaxf(s[2][r], s[3][r]));
        rm = fmaxf(rm, __shfl_xor(rm, 1, 32));
        rm = fmaxf(rm, __shfl_xor(rm, 2, 32));
        rm = fmaxf(rm, __shfl_xor(rm, 4, 32));
        rm = fmaxf(rm, __shfl_xor(rm, 8, 32));
        float mn    = fmaxf(m_run[r], rm);
        float alpha = exp2f(m_run[r] - mn);
        m_run[r] = mn;
        float rs = 0.0f;
#pragma unroll
        for (int t = 0; t < 4; ++t) {
          float p = exp2f(s[t][r] - mn);
          s[t][r] = p;
          rs += p;
        }
        rs += __shfl_xor(rs, 1, 32);
        rs += __shfl_xor(rs, 2, 32);
        rs += __shfl_xor(rs, 4, 32);
        rs += __shfl_xor(rs, 8, 32);
        l_run[r] = l_run[r] * alpha + rs;
#pragma unroll
        for (int t = 0; t < 4; ++t) o[t][r] *= alpha;
      }
      // ---- P (C-layout) -> LDS -> A-layout fragments ----
#pragma unroll
      for (int t = 0; t < 4; ++t)
#pragma unroll
        for (int r = 0; r < 8; ++r)
          Pbuf[wave][(mloc + r) * 64 + t * 16 + ncol] = (_Float16)s[t][r];
      asm volatile("s_wait_dscnt 0x0" ::: "memory");
      v16h pf0 = load_a_frag(&Pbuf[wave][0], 64, 0, 0, lane);
      v16h pf1 = load_a_frag(&Pbuf[wave][0], 64, 0, 32, lane);
      // ---- O += P V ----
#pragma unroll
      for (int t = 0; t < 4; ++t) {
        v16h vf0 = load_b_frag(&Vbuf[buf][0], 64, t * 16, 0, lane);
        o[t] = wmma_f16(pf0, vf0, o[t]);
        v16h vf1 = load_b_frag(&Vbuf[buf][0], 64, t * 16, 32, lane);
        o[t] = wmma_f16(pf1, vf1, o[t]);
      }
    }
    __syncthreads();
  }

  // ---- epilogue: normalize, store O f16 [b][l][h*HD + d] ----
#pragma unroll
  for (int t = 0; t < 4; ++t) {
#pragma unroll
    for (int r = 0; r < 8; ++r) {
      float v = o[t][r] / l_run[r];
      int l   = qrow0 + mloc + r;
      int col = h * HD_ + t * 16 + ncol;
      O[(((size_t)b * L_) + l) * (H_ * HD_) + col] = (_Float16)v;
    }
  }
}

// ---------------------------------------------------------------------------
// Launch
// ---------------------------------------------------------------------------
extern "C" void kernel_launch(void* const* d_in, const int* in_sizes, int n_in,
                              void* d_out, int out_size, void* d_ws, size_t ws_size,
                              hipStream_t stream) {
  const float* x  = (const float*)d_in[0];
  const float* fc = (const float*)d_in[1];
  const float* wq = (const float*)d_in[2];
  const float* wk = (const float*)d_in[3];
  const float* wv = (const float*)d_in[4];
  const float* wo = (const float*)d_in[5];
  float* out = (float*)d_out;

  const int nX  = B_ * L_ * DIM_;
  const int nWq = H_ * HD_ * DIM_;
  const int nWk = KV_ * HD_ * DIM_;
  const int nQ  = B_ * H_ * L_ * HD_;
  const int nK  = B_ * KV_ * L_ * HD_;

  _Float16* base = (_Float16*)d_ws;
  _Float16* xh  = base;
  _Float16* wqh = xh  + nX;
  _Float16* wkh = wqh + nWq;
  _Float16* wvh = wkh + nWk;
  _Float16* woh = wvh + nWk;
  _Float16* Qd  = woh + nWq;
  _Float16* Kd  = Qd  + nQ;
  _Float16* Vd  = Kd  + nK;
  _Float16* Od  = Vd  + nK;

  cvt_f32_f16<<<(nX  + 255) / 256, 256, 0, stream>>>(x,  xh,  nX);
  cvt_f32_f16<<<(nWq + 255) / 256, 256, 0, stream>>>(wq, wqh, nWq);
  cvt_f32_f16<<<(nWk + 255) / 256, 256, 0, stream>>>(wk, wkh, nWk);
  cvt_f32_f16<<<(nWk + 255) / 256, 256, 0, stream>>>(wv, wvh, nWk);
  cvt_f32_f16<<<(nWq + 255) / 256, 256, 0, stream>>>(wo, woh, nWq);

  const int M = B_ * L_;
  gemm_wmma<<<dim3((H_ * HD_) / 128, M / 128), 128, 0, stream>>>(
      xh, wqh, Qd, nullptr, M, H_ * HD_, DIM_, 0);
  gemm_wmma<<<dim3((KV_ * HD_) / 128, M / 128), 128, 0, stream>>>(
      xh, wkh, Kd, nullptr, M, KV_ * HD_, DIM_, 1);
  gemm_wmma<<<dim3((KV_ * HD_) / 128, M / 128), 128, 0, stream>>>(
      xh, wvh, Vd, nullptr, M, KV_ * HD_, DIM_, 2);

  rope_kernel<<<(B_ * H_ * L_ * 32) / 256, 256, 0, stream>>>(Qd, fc, B_ * H_ * L_ * 32);
  rope_kernel<<<(B_ * KV_ * L_ * 32) / 256, 256, 0, stream>>>(Kd, fc, B_ * KV_ * L_ * 32);

  flash_attn<<<dim3(L_ / 128, B_ * H_), 256, 0, stream>>>(Qd, Kd, Vd, Od);

  gemm_wmma<<<dim3(DIM_ / 128, M / 128), 128, 0, stream>>>(
      Od, woh, nullptr, out, M, DIM_, H_ * HD_, 3);
}
